// MoeHashLayer_40853728919571
// MI455X (gfx1250) — compile-verified, matrix operand order checked by reading, not run on hardware
//
#include <hip/hip_runtime.h>

typedef __attribute__((ext_vector_type(16))) _Float16 v16h;
typedef __attribute__((ext_vector_type(8)))  _Float16 v8h;
typedef __attribute__((ext_vector_type(8)))  float    v8f;

constexpr int DIM  = 512;
constexpr int HID  = 2048;
constexpr int NE   = 8;
constexpr int NTOK = 4096;                 // B*T
constexpr int CHUNK  = 256;                // hidden cols per chunk
constexpr int NCHUNK = HID / CHUNK;        // 8
constexpr int XSTRIDE = DIM + 8;           // padded LDS stride (halves)
constexpr int HSTRIDE = CHUNK + 8;

// packed-weight geometry: tiles of 32(K) x 16(N), 512 halfs (1024 B) each, fragment-major
constexpr int  GI_TILES = (HID / 16) * (DIM / 32);   // 2048 tiles/expert (w_gate, w_in)
constexpr int  O_TILES  = (DIM / 16) * (HID / 32);   // 2048 tiles/expert (w_out)
constexpr long GI_HALFS = (long)GI_TILES * 512;      // 1 Mi halfs / expert
constexpr long O_HALFS  = (long)O_TILES  * 512;

#define WMMA_F16(A, B, C) __builtin_amdgcn_wmma_f32_16x16x32_f16(false, A, false, B, (short)0, C, false, false)

// k covered by half-element i of a 16-bit A/B fragment (ISA 7.12.2, wave32):
// VGPR j<4: k=2j,2j+1 ; VGPR j>=4: k=2j+8,2j+9 ; +8 for lanes 16..31
__device__ __forceinline__ int frag_k(int i, int laneHi) {
    return (i < 8 ? i : i + 8) + laneHi * 8;
}

__device__ __forceinline__ float silu_mul(float g, float i) {
    return g * __builtin_amdgcn_rcpf(1.0f + __expf(-g)) * i;   // v_rcp_f32, no IEEE div
}

__device__ __forceinline__ v16h load_a(const _Float16* p) {  // LDS: k..k+7 | k+16..k+23
    v8h lo = *(const v8h*)p;
    v8h hi = *(const v8h*)(p + 16);
    return __builtin_shufflevector(lo, hi, 0,1,2,3,4,5,6,7,8,9,10,11,12,13,14,15);
}

// tbase is wave-uniform (SGPR pair); laneOff is the only divergent part (32-bit VGPR)
// -> global_load_b128 vdst, voff, s[b:b+1] offset:imm
__device__ __forceinline__ v16h load_tile(const _Float16* tbase, unsigned laneOff) {
    const char* p = (const char*)tbase + laneOff;
    v8h lo = *(const v8h*)p;
    v8h hi = *(const v8h*)(p + 16);
    return __builtin_shufflevector(lo, hi, 0,1,2,3,4,5,6,7,8,9,10,11,12,13,14,15);
}

__global__ void moe_zero_counts(int* counts) {
    if (threadIdx.x < NE) counts[threadIdx.x] = 0;
}

__global__ void moe_gather(const int* __restrict__ route, int* __restrict__ counts,
                           int* __restrict__ lists) {
    int i = blockIdx.x * blockDim.x + threadIdx.x;
    if (i < NTOK) {
        int e = route[i];
        int slot = atomicAdd(&counts[e], 1);
        lists[e * NTOK + slot] = i;
    }
}

// Repack f32 W[e][K][N] (row-major) -> f16 fragment-major tiles. One wave per 32x16 tile.
__global__ void pack_w(const float* __restrict__ W, _Float16* __restrict__ P,
                       int K, int N) {
    int waveId = (blockIdx.x * blockDim.x + threadIdx.x) >> 5;
    int lane   = threadIdx.x & 31;
    int ksteps = K / 32;
    int tilesPerE = (N / 16) * ksteps;
    int e = waveId / tilesPerE;
    if (e >= NE) return;
    int t     = waveId % tilesPerE;
    int ntile = t / ksteps;
    int ks    = t % ksteps;
    const float* src = W + (long)e * K * N;
    int n      = ntile * 16 + (lane & 15);
    int laneHi = lane >> 4;
    v16h r;
    #pragma unroll
    for (int i = 0; i < 16; ++i) {
        unsigned k = ks * 32 + frag_k(i, laneHi);
        r[i] = (_Float16)src[(unsigned)(k * N + n)];   // coalesced: 16 lanes span 64B/row
    }
    _Float16* dst = P + (((long)e * tilesPerE + t) * 32 + lane) * 16;
    *(v8h*)dst       = __builtin_shufflevector(r, r, 0,1,2,3,4,5,6,7);
    *(v8h*)(dst + 8) = __builtin_shufflevector(r, r, 8,9,10,11,12,13,14,15);
}

// ---------------- packed-weight FFN: 32 tokens/block, 8 waves ----------------
__launch_bounds__(256)
__global__ void moe_ffn_packed(const float* __restrict__ x,
                               const _Float16* __restrict__ pWg,
                               const _Float16* __restrict__ pWi,
                               const _Float16* __restrict__ pWo,
                               const float* __restrict__ b_in,
                               const float* __restrict__ b_gate,
                               const float* __restrict__ b_out,
                               const int* __restrict__ counts,
                               const int* __restrict__ lists,
                               float* __restrict__ out) {
    __shared__ _Float16 Xs[32 * XSTRIDE];   // 32x512 f16
    __shared__ _Float16 Hs[32 * HSTRIDE];   // 32x256 f16 hidden chunk
    __shared__ int      toks[32];

    const int e    = blockIdx.x >> 7;       // 128 tiles/expert
    const int tile = blockIdx.x & 127;
    const int cnt  = counts[e];
    if (tile * 32 >= cnt) return;           // block-uniform exit (EXEC all-1s at WMMA)

    const int tid       = threadIdx.x;
    const unsigned lane = tid & 31;
    // tid>>5 is constant within a wave32; readfirstlane makes that provable -> SGPR bases
    const int wave   = __builtin_amdgcn_readfirstlane(tid >> 5);
    const int laneLo = lane & 15;
    const int laneHi = lane >> 4;
    const unsigned laneOff = lane * 32u;    // per-lane byte offset inside a packed tile

    if (tid < 32) {
        int g = tile * 32 + tid;
        toks[tid] = (g < cnt) ? lists[e * NTOK + g] : -1;
    }
    __syncthreads();

    {   // gather+convert X tile: 8 threads/row, 64 cols each
        int row = tid >> 3;
        int cb  = (tid & 7) * 64;
        int tok = toks[row];
        const float* src = x + (long)tok * DIM + cb;
        for (int j = 0; j < 64; j += 4) {
            float4 v4 = make_float4(0.f, 0.f, 0.f, 0.f);
            if (tok >= 0) v4 = *(const float4*)(src + j);
            _Float16* d = &Xs[row * XSTRIDE + cb + j];
            d[0] = (_Float16)v4.x; d[1] = (_Float16)v4.y;
            d[2] = (_Float16)v4.z; d[3] = (_Float16)v4.w;
        }
    }
    __syncthreads();

    const _Float16* baseWg = pWg + (long)e * GI_HALFS;   // uniform -> SGPR pair
    const _Float16* baseWi = pWi + (long)e * GI_HALFS;
    const _Float16* baseWo = pWo + (long)e * O_HALFS;

    v8f accO[8] = {};   // 32 tokens x 64 out cols per wave, f32, lives across chunks

    for (int c = 0; c < NCHUNK; ++c) {
        // ---- phase 1: hidden chunk cols [c*256, c*256+256), wave covers 32 cols ----
        for (int sub = 0; sub < 2; ++sub) {
            const int colBase = wave * 32 + sub * 16;             // wave-uniform
            const unsigned ntile = (unsigned)(c * CHUNK + colBase) >> 4;
            const _Float16* tgG = baseWg + (size_t)ntile * 16u * 512u;  // uniform tile base
            const _Float16* tgI = baseWi + (size_t)ntile * 16u * 512u;
            v8f aG0 = {}, aG1 = {}, aI0 = {}, aI1 = {};
            for (unsigned ks = 0; ks < DIM / 32; ++ks) {
                v16h bg = load_tile(tgG + ks * 512u, laneOff);   // saddr + imm form
                v16h bi = load_tile(tgI + ks * 512u, laneOff);
                v16h a0 = load_a(&Xs[laneLo * XSTRIDE + ks * 32 + laneHi * 8]);
                v16h a1 = load_a(&Xs[(16 + laneLo) * XSTRIDE + ks * 32 + laneHi * 8]);
                aG0 = WMMA_F16(a0, bg, aG0);
                aI0 = WMMA_F16(a0, bi, aI0);
                aG1 = WMMA_F16(a1, bg, aG1);
                aI1 = WMMA_F16(a1, bi, aI1);
            }
            const unsigned gcol = (unsigned)(e * HID + c * CHUNK + colBase + laneLo);
            const float bgv = b_gate[gcol];
            const float biv = b_in [gcol];
            #pragma unroll
            for (int v = 0; v < 8; ++v) {
                int m = v + laneHi * 8;
                Hs[m * HSTRIDE + colBase + laneLo] =
                    (_Float16)silu_mul(aG0[v] + bgv, aI0[v] + biv);
                Hs[(m + 16) * HSTRIDE + colBase + laneLo] =
                    (_Float16)silu_mul(aG1[v] + bgv, aI1[v] + biv);
            }
        }
        __syncthreads();
        // ---- phase 2: out += H_chunk @ Wo[chunk rows], wave covers 64 out cols ----
        // uniform base for this wave+chunk; t,ks become immediate offsets after unroll
        const _Float16* tgO = baseWo + ((size_t)(wave * 4) * 64u + (size_t)c * 8u) * 512u;
        for (unsigned ks = 0; ks < CHUNK / 32; ++ks) {
            v16h a0 = load_a(&Hs[laneLo * HSTRIDE + ks * 32 + laneHi * 8]);
            v16h a1 = load_a(&Hs[(16 + laneLo) * HSTRIDE + ks * 32 + laneHi * 8]);
            #pragma unroll
            for (unsigned t = 0; t < 4; ++t) {
                v16h b = load_tile(tgO + (t * 64u + ks) * 512u, laneOff);
                accO[t]     = WMMA_F16(a0, b, accO[t]);
                accO[4 + t] = WMMA_F16(a1, b, accO[4 + t]);
            }
        }
        __syncthreads();   // Hs reused next chunk
    }

    #pragma unroll
    for (int t = 0; t < 4; ++t) {
        const int ncol = wave * 64 + t * 16 + laneLo;
        const float bo = b_out[e * DIM + ncol];
        #pragma unroll
        for (int v = 0; v < 8; ++v) {
            int m = v + laneHi * 8;
            int tok0 = toks[m], tok1 = toks[m + 16];
            if (tok0 >= 0) out[(long)tok0 * DIM + ncol] = accO[t][v] + bo;
            if (tok1 >= 0) out[(long)tok1 * DIM + ncol] = accO[4 + t][v] + bo;
        }
    }
}

// ---------------- fallback: direct f32-weight FFN, 16 tokens/block ----------------
__launch_bounds__(256)
__global__ void moe_ffn_direct(const float* __restrict__ x,
                               const float* __restrict__ w_in,  const float* __restrict__ b_in,
                               const float* __restrict__ w_gate,const float* __restrict__ b_gate,
                               const float* __restrict__ w_out, const float* __restrict__ b_out,
                               const int* __restrict__ counts,  const int* __restrict__ lists,
                               float* __restrict__ out) {
    __shared__ _Float16 Xs[16 * XSTRIDE];
    __shared__ _Float16 Hs[16 * HSTRIDE];
    __shared__ int      toks[16];

    const int e    = blockIdx.x >> 8;
    const int tile = blockIdx.x & 255;
    const int cnt  = counts[e];
    if (tile * 16 >= cnt) return;

    const int tid    = threadIdx.x;
    const int lane   = tid & 31;
    const int wave   = __builtin_amdgcn_readfirstlane(tid >> 5);
    const int laneLo = lane & 15;
    const int laneHi = lane >> 4;

    if (tid < 16) {
        int g = tile * 16 + tid;
        toks[tid] = (g < cnt) ? lists[e * NTOK + g] : -1;
    }
    __syncthreads();
    {
        int row = tid >> 4, cb = (tid & 15) * 32;
        int tok = toks[row];
        const float* src = x + (long)tok * DIM + cb;
        for (int j = 0; j < 32; j += 4) {
            float4 v4 = make_float4(0.f, 0.f, 0.f, 0.f);
            if (tok >= 0) v4 = *(const float4*)(src + j);
            _Float16* d = &Xs[row * XSTRIDE + cb + j];
            d[0] = (_Float16)v4.x; d[1] = (_Float16)v4.y;
            d[2] = (_Float16)v4.z; d[3] = (_Float16)v4.w;
        }
    }
    __syncthreads();

    const float* Wg = w_gate + (long)e * DIM * HID;
    const float* Wi = w_in   + (long)e * DIM * HID;
    const float* Wo = w_out  + (long)e * HID * DIM;
    v8f accO[4] = {};

    for (int c = 0; c < NCHUNK; ++c) {
        for (int sub = 0; sub < 2; ++sub) {
            const int ncol = wave * 32 + sub * 16 + laneLo;
            const unsigned gcol = (unsigned)(c * CHUNK + ncol);
            v8f accG = {}, accI = {};
            for (int ks = 0; ks < DIM / 32; ++ks) {
                v16h a = load_a(&Xs[laneLo * XSTRIDE + ks * 32 + laneHi * 8]);
                v16h bg, bi;
                #pragma unroll
                for (int i = 0; i < 16; ++i) {
                    unsigned k = (unsigned)(ks * 32 + frag_k(i, laneHi));
                    bg[i] = (_Float16)Wg[k * (unsigned)HID + gcol];
                    bi[i] = (_Float16)Wi[k * (unsigned)HID + gcol];
                }
                accG = WMMA_F16(a, bg, accG);
                accI = WMMA_F16(a, bi, accI);
            }
            const float bgv = b_gate[e * HID + gcol];
            const float biv = b_in [e * HID + gcol];
            #pragma unroll
            for (int v = 0; v < 8; ++v) {
                int m = v + laneHi * 8;
                Hs[m * HSTRIDE + ncol] = (_Float16)silu_mul(accG[v] + bgv, accI[v] + biv);
            }
        }
        __syncthreads();
        for (int ks = 0; ks < CHUNK / 32; ++ks) {
            v16h a = load_a(&Hs[laneLo * HSTRIDE + ks * 32 + laneHi * 8]);
            #pragma unroll
            for (int t = 0; t < 4; ++t) {
                const int ncol = wave * 64 + t * 16 + laneLo;
                v16h b;
                #pragma unroll
                for (int i = 0; i < 16; ++i) {
                    unsigned k = (unsigned)(c * CHUNK + ks * 32 + frag_k(i, laneHi));
                    b[i] = (_Float16)Wo[k * (unsigned)DIM + (unsigned)ncol];
                }
                accO[t] = WMMA_F16(a, b, accO[t]);
            }
        }
        __syncthreads();
    }
    #pragma unroll
    for (int t = 0; t < 4; ++t) {
        const int ncol = wave * 64 + t * 16 + laneLo;
        const float bo = b_out[e * DIM + ncol];
        #pragma unroll
        for (int v = 0; v < 8; ++v) {
            int m = v + laneHi * 8;
            int tok = toks[m];
            if (tok >= 0) out[(long)tok * DIM + ncol] = accO[t][v] + bo;
        }
    }
}

extern "C" void kernel_launch(void* const* d_in, const int* in_sizes, int n_in,
                              void* d_out, int out_size, void* d_ws, size_t ws_size,
                              hipStream_t stream) {
    const float* x      = (const float*)d_in[0];
    const int*   route  = (const int*)  d_in[1];
    const float* w_in   = (const float*)d_in[2];
    const float* b_in   = (const float*)d_in[3];
    const float* w_gate = (const float*)d_in[4];
    const float* b_gate = (const float*)d_in[5];
    const float* w_out  = (const float*)d_in[6];
    const float* b_out  = (const float*)d_in[7];
    float* out = (float*)d_out;

    int* counts = (int*)d_ws;
    int* lists  = (int*)((char*)d_ws + 64);
    const size_t packOff = 131328;  // 64 + NE*NTOK*4 rounded up to 256
    const size_t giBytes = (size_t)NE * GI_HALFS * 2;     // 16 MiB
    const size_t oBytes  = (size_t)NE * O_HALFS  * 2;     // 16 MiB
    const size_t need    = packOff + 2 * giBytes + oBytes;

    moe_zero_counts<<<1, 32, 0, stream>>>(counts);
    moe_gather<<<NTOK / 256, 256, 0, stream>>>(route, counts, lists);

    if (ws_size >= need) {
        _Float16* pWg = (_Float16*)((char*)d_ws + packOff);
        _Float16* pWi = pWg + (size_t)NE * GI_HALFS;
        _Float16* pWo = pWi + (size_t)NE * GI_HALFS;
        pack_w<<<2048, 256, 0, stream>>>(w_gate, pWg, DIM, HID);
        pack_w<<<2048, 256, 0, stream>>>(w_in,   pWi, DIM, HID);
        pack_w<<<2048, 256, 0, stream>>>(w_out,  pWo, HID, DIM);
        moe_ffn_packed<<<NE * 128, 256, 0, stream>>>(x, pWg, pWi, pWo,
                                                     b_in, b_gate, b_out,
                                                     counts, lists, out);
    } else {
        moe_ffn_direct<<<NE * 256, 256, 0, stream>>>(x, w_in, b_in, w_gate, b_gate,
                                                     w_out, b_out, counts, lists, out);
    }
}
